// TACA_60790967107664
// MI455X (gfx1250) — matrix operand order repaired and checked by reading, not compiled, not run on hardware
//
#include <hip/hip_runtime.h>
#include <hip/hip_bf16.h>

// Problem constants (match reference)
#define BB 4
#define NN 4096
#define LL 256
#define DD 2048
#define HH 16
#define HD 128

typedef __attribute__((ext_vector_type(16))) _Float16 v16h;
typedef __attribute__((ext_vector_type(8)))  _Float16 v8h;
typedef __attribute__((ext_vector_type(4)))  _Float16 v4h;
typedef __attribute__((ext_vector_type(8)))  float    v8f;
typedef unsigned int v4u __attribute__((ext_vector_type(4)));
typedef int          v8i __attribute__((ext_vector_type(8)));
typedef int          v4i __attribute__((ext_vector_type(4)));

#define WMMA_F32_F16(a, b, c) \
  __builtin_amdgcn_wmma_f32_16x16x32_f16(false, (a), false, (b), (short)0, (c), false, false)

#if __has_builtin(__builtin_amdgcn_tensor_load_to_lds)
#define HAVE_TDM 1
#if __clang_major__ >= 23
#define TDM_6ARG 1
#endif
#endif

// Async global->LDS 16-byte copy (GLOBAL_LOAD_ASYNC_TO_LDS_B128, ASYNCcnt)
__device__ __forceinline__ void async_ld_b128(void* lds, const void* gptr) {
  uint32_t loff = (uint32_t)(size_t)lds;  // low 32 bits of flat LDS addr = LDS offset
  asm volatile("global_load_async_to_lds_b128 %0, %1, off"
               :: "v"(loff), "v"(gptr) : "memory");
}
__device__ __forceinline__ void wait_async0() {
  asm volatile("s_wait_asynccnt 0" ::: "memory");
}
__device__ __forceinline__ void wait_tensor0() {
#if __has_builtin(__builtin_amdgcn_s_wait_tensorcnt)
  __builtin_amdgcn_s_wait_tensorcnt(0);
#else
  asm volatile("s_wait_tensorcnt 0" ::: "memory");
#endif
}

#ifdef HAVE_TDM
// TDM 2D tile load: rows x 32 f16 tile from row-major (ld=K) global into packed
// row-major (ld=32) LDS.  D# layout per CDNA5 ISA ch.8 (tensor descriptor).
__device__ __forceinline__ void tdm_load_2d(uint32_t lds_off, const void* gaddr,
                                            uint32_t rows, uint32_t K) {
  uint64_t ga = (uint64_t)gaddr;
  v4u g0;
  g0[0] = 1u;                                   // count=1, user mode, no gather
  g0[1] = lds_off;                              // lds_addr (bytes)
  g0[2] = (uint32_t)ga;                         // global_addr[31:0]
  g0[3] = (uint32_t)(ga >> 32) | (2u << 30);    // global_addr[56:32] | type=2
  v8i g1;
  g1[0] = (int)(1u << 16);                      // data_size=1 (2 bytes)
  g1[1] = (int)((K & 0xFFFFu) << 16);           // tensor_dim0[15:0] @ bits 63:48
  g1[2] = (int)((K >> 16) | ((rows & 0xFFFFu) << 16));  // dim0 hi | tensor_dim1 lo
  g1[3] = (int)((rows >> 16) | (32u << 16));    // tensor_dim1 hi | tile_dim0=32
  g1[4] = (int)(rows & 0xFFFFu);                // tile_dim1=rows, tile_dim2=0
  g1[5] = (int)K;                               // tensor_dim0_stride[31:0]
  g1[6] = 0;                                    // stride hi | dim1_stride lo
  g1[7] = 0;
  v4i z4 = {0, 0, 0, 0};
#ifdef TDM_6ARG
  v8i z8 = {0, 0, 0, 0, 0, 0, 0, 0};
  __builtin_amdgcn_tensor_load_to_lds(g0, g1, z4, z4, z8, 0);
#else
  __builtin_amdgcn_tensor_load_to_lds(g0, g1, z4, z4, 0);
#endif
}
#endif

// A fragment (16x32 f16): lane half grp holds K {grp*8..+7, 16+grp*8..+7}
__device__ __forceinline__ v16h ld_a_frag(const _Float16* rowp, int grp) {
  union { v16h v; v8h h[2]; } u;
  u.h[0] = *(const v8h*)(rowp + grp * 8);
  u.h[1] = *(const v8h*)(rowp + 16 + grp * 8);
  return u.v;
}
// B fragment (32x16 f16): lane holds output column; lane half grp holds K {grp*8..+7, 16+grp*8..+7}
__device__ __forceinline__ v16h ld_b_frag(const _Float16* rowp, int grp) {
  union { v16h v; v8h h[2]; } u;
  u.h[0] = *(const v8h*)(rowp + grp * 8);
  u.h[1] = *(const v8h*)(rowp + 16 + grp * 8);
  return u.v;
}

// ---------------------------------------------------------------------------
// f32 -> f16 conversion pass (bandwidth-bound; enables async/TDM GEMM staging)
// ---------------------------------------------------------------------------
__global__ __launch_bounds__(256)
void cvt_f32_f16(const float* __restrict__ in, _Float16* __restrict__ out, size_t n) {
  size_t i = ((size_t)blockIdx.x * 256 + threadIdx.x) * 8;
  if (i + 8 <= n) {
    float4 a = *(const float4*)(in + i);
    float4 b = *(const float4*)(in + i + 4);
    v8h h;
    h[0] = (_Float16)a.x; h[1] = (_Float16)a.y; h[2] = (_Float16)a.z; h[3] = (_Float16)a.w;
    h[4] = (_Float16)b.x; h[5] = (_Float16)b.y; h[6] = (_Float16)b.z; h[7] = (_Float16)b.w;
    *(v8h*)(out + i) = h;
  }
}

// ---------------------------------------------------------------------------
// Timestep MLP -> per-batch temperature
// ---------------------------------------------------------------------------
__global__ __launch_bounds__(128)
void temp_kernel(const int* __restrict__ ts, const float* __restrict__ w1,
                 const float* __restrict__ b1, const float* __restrict__ w2,
                 const float* __restrict__ b2, float* __restrict__ temp) {
  __shared__ float red[128];
  const int tid = threadIdx.x;
  for (int b = 0; b < BB; ++b) {
    float t = (float)ts[b];
    float acc = b1[tid];
    for (int j = 0; j < 32; ++j) {
      float fr = __expf(-6.9077552789821368f * (float)j / 32.0f);  // -ln(1000)*j/32
      float a = t * fr;
      acc += cosf(a) * w1[tid * 64 + j] + sinf(a) * w1[tid * 64 + 32 + j];
    }
    float h = acc / (1.0f + __expf(-acc));  // silu
    red[tid] = h * w2[tid];
    __syncthreads();
    for (int s = 64; s > 0; s >>= 1) {
      if (tid < s) red[tid] += red[tid + s];
      __syncthreads();
    }
    if (tid == 0) {
      float d = tanhf(red[0] + b2[0]) * 0.5f;
      temp[b] = fmaxf(0.1f, 1.0f + d);
    }
    __syncthreads();
  }
}

// ---------------------------------------------------------------------------
// GEMM: Out[m,n] = sum_k A[m,k]*W[n,k]; A,W f16; optional per-head RMSNorm
// (N tile of 128 == one head).  Block tile 256x128, 256 threads = 8 waves in
// a 4x2 grid, 64x64 per wave (16 ds_load_b128 per 16 WMMAs).
// Staging: double-buffered TDM tensor_load_to_lds (wave 0 issues the DMA,
// TENSORcnt), falling back to per-lane GLOBAL_LOAD_ASYNC_TO_LDS_B128.
// ---------------------------------------------------------------------------
template <typename OutT, bool NORM>
__global__ __launch_bounds__(256)
void gemm_tn(const _Float16* __restrict__ A, const _Float16* __restrict__ W,
             const float* __restrict__ nw, OutT* __restrict__ Out,
             int M, int K, int Nout) {
  __shared__ _Float16 lsA[2][256 * 32];   // 32 KB
  __shared__ _Float16 lsW[2][128 * 32];   // 16 KB
  __shared__ float lsPart[2][256];        //  2 KB

  const int tid = threadIdx.x;
  const int wv = tid >> 5;
  const int lane = tid & 31, l16 = lane & 15, grp = lane >> 4;
  const int wr = wv & 3, wc = wv >> 2;  // 4x2 wave grid
  const int nbase = blockIdx.x * 128;
  const int mbase = blockIdx.y * 256;

  v8f acc[4][4];
#pragma unroll
  for (int mi = 0; mi < 4; ++mi)
#pragma unroll
    for (int ni = 0; ni < 4; ++ni)
#pragma unroll
      for (int r = 0; r < 8; ++r) acc[mi][ni][r] = 0.0f;

  auto stage_tiles = [&](int kcol, int buf) {
#ifdef HAVE_TDM
    if (wv == 0) {
      tdm_load_2d((uint32_t)(size_t)&lsA[buf][0],
                  A + (size_t)mbase * K + kcol, 256u, (uint32_t)K);
      tdm_load_2d((uint32_t)(size_t)&lsW[buf][0],
                  W + (size_t)nbase * K + kcol, 128u, (uint32_t)K);
    }
#else
#pragma unroll
    for (int j = 0; j < 4; ++j) {
      int lin = j * 256 + tid;
      int r = lin >> 2, cb = (lin & 3) * 8;
      async_ld_b128(&lsA[buf][r * 32 + cb], A + (size_t)(mbase + r) * K + kcol + cb);
    }
#pragma unroll
    for (int j = 0; j < 2; ++j) {
      int lin = j * 256 + tid;
      int r = lin >> 2, cb = (lin & 3) * 8;
      async_ld_b128(&lsW[buf][r * 32 + cb], W + (size_t)(nbase + r) * K + kcol + cb);
    }
#endif
  };
  auto stage_wait = [&]() {
#ifdef HAVE_TDM
    if (wv == 0) wait_tensor0();
#else
    wait_async0();
#endif
    __syncthreads();
  };

  const int nk = K >> 5;
  stage_tiles(0, 0);
  stage_wait();

  for (int kt = 0; kt < nk; ++kt) {
    int cur = kt & 1;
    if (kt + 1 < nk) stage_tiles((kt + 1) * 32, 1 - cur);
    v16h bfr[4];
#pragma unroll
    for (int ni = 0; ni < 4; ++ni)
      bfr[ni] = ld_b_frag(&lsW[cur][(wc * 64 + ni * 16 + l16) * 32], grp);
#pragma unroll
    for (int mi = 0; mi < 4; ++mi) {
      v16h afr = ld_a_frag(&lsA[cur][(wr * 64 + mi * 16 + l16) * 32], grp);
#pragma unroll
      for (int ni = 0; ni < 4; ++ni)
        acc[mi][ni] = WMMA_F32_F16(afr, bfr[ni], acc[mi][ni]);
    }
    stage_wait();  // fills of the other buffer must land before barrier
  }

  if constexpr (NORM) {
#pragma unroll
    for (int mi = 0; mi < 4; ++mi)
#pragma unroll
      for (int r = 0; r < 8; ++r) {
        float sq = 0.0f;
#pragma unroll
        for (int ni = 0; ni < 4; ++ni) sq += acc[mi][ni][r] * acc[mi][ni][r];
        sq += __shfl_xor(sq, 1); sq += __shfl_xor(sq, 2);
        sq += __shfl_xor(sq, 4); sq += __shfl_xor(sq, 8);
        if (l16 == 0) lsPart[wc][wr * 64 + mi * 16 + grp * 8 + r] = sq;
      }
    __syncthreads();
    float wgt[4];
#pragma unroll
    for (int ni = 0; ni < 4; ++ni) wgt[ni] = nw[wc * 64 + ni * 16 + l16];
#pragma unroll
    for (int mi = 0; mi < 4; ++mi)
#pragma unroll
      for (int r = 0; r < 8; ++r) {
        int row = wr * 64 + mi * 16 + grp * 8 + r;
        float tot = lsPart[0][row] + lsPart[1][row];
        float rs = rsqrtf(tot * (1.0f / 128.0f) + 1e-6f);
        size_t base = (size_t)(mbase + row) * Nout + nbase + wc * 64;
#pragma unroll
        for (int ni = 0; ni < 4; ++ni)
          Out[base + ni * 16 + l16] = (OutT)(acc[mi][ni][r] * rs * wgt[ni]);
      }
  } else {
#pragma unroll
    for (int mi = 0; mi < 4; ++mi)
#pragma unroll
      for (int r = 0; r < 8; ++r) {
        int row = wr * 64 + mi * 16 + grp * 8 + r;
        size_t base = (size_t)(mbase + row) * Nout + nbase + wc * 64;
#pragma unroll
        for (int ni = 0; ni < 4; ++ni)
          Out[base + ni * 16 + l16] = (OutT)acc[mi][ni][r];
      }
  }
}

// ---------------------------------------------------------------------------
// Fused cross-attention with online softmax.
// Block = (64 queries, head h, batch b), 4 waves. L=256 keys in 4 chunks of 64.
// Q/K tiles staged with async-to-LDS; V staged transposed via VGPR path.
// ---------------------------------------------------------------------------
__global__ __launch_bounds__(128)
void attn_kernel(const _Float16* __restrict__ Qm, const _Float16* __restrict__ Km,
                 const _Float16* __restrict__ Vm, const float* __restrict__ temp,
                 const float* __restrict__ imb, _Float16* __restrict__ Om) {
  __shared__ _Float16 lsQ[64 * 128];   // 16 KB, queries (row-major, ld=128)
  __shared__ _Float16 lsK[64 * 128];   // 16 KB, key chunk (row = key, ld=128)
  __shared__ _Float16 lsVt[128 * 64];  // 16 KB, V chunk transposed (row = d, ld=64)
  __shared__ _Float16 lsP[64 * 64];    //  8 KB, probabilities (row = query, ld=64)

  const int tid = threadIdx.x;
  const int w = tid >> 5;
  const int lane = tid & 31, l16 = lane & 15, grp = lane >> 4;
  const int qb = blockIdx.x * 64;
  const int h = blockIdx.y;
  const int b = blockIdx.z;

  const float scl = 0.08838834764831845f * imb[h] / temp[b];  // isqrt(128)*imb/temp

  // Async-load query tile once
#pragma unroll
  for (int i = 0; i < 8; ++i) {
    int lin = i * 128 + tid;
    int r = lin >> 4;
    int c = (lin & 15) << 3;
    async_ld_b128(&lsQ[r * 128 + c],
                  Qm + (size_t)(b * NN + qb + r) * DD + h * HD + c);
  }

  float mst[8], sst[8];
  v8f o[8];
#pragma unroll
  for (int r = 0; r < 8; ++r) { mst[r] = -3.0e38f; sst[r] = 0.0f; }
#pragma unroll
  for (int j = 0; j < 8; ++j)
#pragma unroll
    for (int r = 0; r < 8; ++r) o[j][r] = 0.0f;

  for (int kc = 0; kc < 4; ++kc) {
    const int l0 = kc * 64;
    // K chunk: async row-major copy. V chunk: transpose through VGPRs.
#pragma unroll
    for (int i = 0; i < 8; ++i) {
      int lin = i * 128 + tid;
      int r = lin >> 4;
      int c = (lin & 15) << 3;
      async_ld_b128(&lsK[r * 128 + c],
                    Km + (size_t)(b * LL + l0 + r) * DD + h * HD + c);
      v8h vd = *(const v8h*)(Vm + (size_t)(b * LL + l0 + r) * DD + h * HD + c);
#pragma unroll
      for (int q = 0; q < 8; ++q) lsVt[(c + q) * 64 + r] = vd[q];
    }
    wait_async0();
    __syncthreads();

    // S = Q * K^T   (wave w owns query rows w*16..w*16+15, all 64 key cols)
    v8f s[4];
#pragma unroll
    for (int ni = 0; ni < 4; ++ni)
#pragma unroll
      for (int r = 0; r < 8; ++r) s[ni][r] = 0.0f;
#pragma unroll
    for (int kt = 0; kt < 4; ++kt) {
      v16h afr = ld_a_frag(&lsQ[(w * 16 + l16) * 128 + kt * 32], grp);
      v16h bfr[4];
#pragma unroll
      for (int ni = 0; ni < 4; ++ni)
        bfr[ni] = ld_b_frag(&lsK[(ni * 16 + l16) * 128 + kt * 32], grp);
#pragma unroll
      for (int ni = 0; ni < 4; ++ni) s[ni] = WMMA_F32_F16(afr, bfr[ni], s[ni]);
    }

    // Online softmax update; write P (f16) to LDS
#pragma unroll
    for (int r = 0; r < 8; ++r) {
      float sv[4];
      float mx = -3.0e38f;
#pragma unroll
      for (int ni = 0; ni < 4; ++ni) {
        sv[ni] = s[ni][r] * scl;
        mx = fmaxf(mx, sv[ni]);
      }
      mx = fmaxf(mx, __shfl_xor(mx, 1));
      mx = fmaxf(mx, __shfl_xor(mx, 2));
      mx = fmaxf(mx, __shfl_xor(mx, 4));
      mx = fmaxf(mx, __shfl_xor(mx, 8));
      float mnew = fmaxf(mst[r], mx);
      float corr = __expf(mst[r] - mnew);
      float ps = 0.0f;
      int prow = w * 16 + grp * 8 + r;
#pragma unroll
      for (int ni = 0; ni < 4; ++ni) {
        float p = __expf(sv[ni] - mnew);
        ps += p;
        lsP[prow * 64 + ni * 16 + l16] = (_Float16)p;
      }
      ps += __shfl_xor(ps, 1);
      ps += __shfl_xor(ps, 2);
      ps += __shfl_xor(ps, 4);
      ps += __shfl_xor(ps, 8);
      sst[r] = sst[r] * corr + ps;
      mst[r] = mnew;
#pragma unroll
      for (int j = 0; j < 8; ++j) o[j][r] *= corr;
    }

    // O += P * V  (P rows are intra-wave; LDS ops are in-order within a wave)
#pragma unroll
    for (int kt = 0; kt < 2; ++kt) {
      v16h afr = ld_a_frag(&lsP[(w * 16 + l16) * 64 + kt * 32], grp);
#pragma unroll
      for (int jj = 0; jj < 8; jj += 4) {
        v16h bfr[4];
#pragma unroll
        for (int j = 0; j < 4; ++j)
          bfr[j] = ld_b_frag(&lsVt[((jj + j) * 16 + l16) * 64 + kt * 32], grp);
#pragma unroll
        for (int j = 0; j < 4; ++j)
          o[jj + j] = WMMA_F32_F16(afr, bfr[j], o[jj + j]);
      }
    }
    __syncthreads();  // protect lsK/lsVt before next chunk's loads
  }

  // Normalize and store
#pragma unroll
  for (int r = 0; r < 8; ++r) {
    float inv = 1.0f / sst[r];
    int row = qb + w * 16 + grp * 8 + r;
#pragma unroll
    for (int j = 0; j < 8; ++j) {
      Om[(size_t)(b * NN + row) * DD + h * HD + j * 16 + l16] =
          (_Float16)(o[j][r] * inv);
    }
  }
}

// ---------------------------------------------------------------------------
extern "C" void kernel_launch(void* const* d_in, const int* in_sizes, int n_in,
                              void* d_out, int out_size, void* d_ws, size_t ws_size,
                              hipStream_t stream) {
  const float* x        = (const float*)d_in[0];
  const float* text_emb = (const float*)d_in[1];
  const int*   timestep = (const int*)d_in[2];
  const float* Wq   = (const float*)d_in[3];
  const float* Wk   = (const float*)d_in[4];
  const float* Wv   = (const float*)d_in[5];
  const float* Wo   = (const float*)d_in[6];
  const float* qn_w = (const float*)d_in[7];
  const float* kn_w = (const float*)d_in[8];
  const float* imb  = (const float*)d_in[9];
  const float* t_w1 = (const float*)d_in[10];
  const float* t_b1 = (const float*)d_in[11];
  const float* t_w2 = (const float*)d_in[12];
  const float* t_b2 = (const float*)d_in[13];

  const size_t nX  = (size_t)BB * NN * DD;  // 33.5M
  const size_t nTE = (size_t)BB * LL * DD;  // 2.1M
  const size_t nW  = (size_t)DD * DD;       // 4.2M

  _Float16* xh  = (_Float16*)d_ws;          // f16 copies of inputs/weights
  _Float16* teh = xh + nX;
  _Float16* wqh = teh + nTE;
  _Float16* wkh = wqh + nW;
  _Float16* wvh = wkh + nW;
  _Float16* woh = wvh + nW;
  _Float16* qws = woh + nW;                 // (B*N, D) rms-normed Q
  _Float16* kws = qws + nX;                 // (B*L, D) rms-normed K
  _Float16* vws = kws + nTE;                // (B*L, D) V
  _Float16* aws = vws + nTE;                // (B*N, D) attention out
  float*    tws = (float*)(aws + nX);       // (B,) temperature

  temp_kernel<<<1, 128, 0, stream>>>(timestep, t_w1, t_b1, t_w2, t_b2, tws);

  cvt_f32_f16<<<(int)(nX  / 2048), 256, 0, stream>>>(x, xh, nX);
  cvt_f32_f16<<<(int)(nTE / 2048), 256, 0, stream>>>(text_emb, teh, nTE);
  cvt_f32_f16<<<(int)(nW  / 2048), 256, 0, stream>>>(Wq, wqh, nW);
  cvt_f32_f16<<<(int)(nW  / 2048), 256, 0, stream>>>(Wk, wkh, nW);
  cvt_f32_f16<<<(int)(nW  / 2048), 256, 0, stream>>>(Wv, wvh, nW);
  cvt_f32_f16<<<(int)(nW  / 2048), 256, 0, stream>>>(Wo, woh, nW);

  gemm_tn<_Float16, true><<<dim3(DD / 128, (BB * NN) / 256), 256, 0, stream>>>(
      xh, wqh, qn_w, qws, BB * NN, DD, DD);
  gemm_tn<_Float16, true><<<dim3(DD / 128, (BB * LL) / 256), 256, 0, stream>>>(
      teh, wkh, kn_w, kws, BB * LL, DD, DD);
  gemm_tn<_Float16, false><<<dim3(DD / 128, (BB * LL) / 256), 256, 0, stream>>>(
      teh, wvh, nullptr, vws, BB * LL, DD, DD);

  attn_kernel<<<dim3(NN / 64, HH, BB), 128, 0, stream>>>(qws, kws, vws, tws, imb, aws);

  gemm_tn<float, false><<<dim3(DD / 128, (BB * NN) / 256), 256, 0, stream>>>(
      aws, woh, nullptr, (float*)d_out, BB * NN, DD, DD);
}
